// GPT_50500225466510
// MI455X (gfx1250) — matrix operand order repaired and checked by reading, not compiled, not run on hardware
//
#include <hip/hip_runtime.h>

#define E_ 1024
#define H_ 16
#define HD_ 64
#define B_ 8
#define TID_ 512
#define TSTIM_ 1024

typedef __bf16 bf16_t;
typedef __bf16 v8bf  __attribute__((ext_vector_type(8)));
typedef __bf16 v16bf __attribute__((ext_vector_type(16)));
typedef float  v8f   __attribute__((ext_vector_type(8)));

// ---------------- device helpers ----------------
__device__ __forceinline__ bf16_t f2bf(float f) {
  unsigned u = __builtin_bit_cast(unsigned, f);
  unsigned r = (u + 0x7FFFu + ((u >> 16) & 1u)) >> 16;   // RNE
  unsigned short s = (unsigned short)r;
  return __builtin_bit_cast(bf16_t, s);
}

// force global (AS1) addressing so loads lower to global_load_b128, not flat_load
typedef const __attribute__((address_space(1))) v8bf*   gv8p;
typedef const __attribute__((address_space(1))) float*  gfp;
typedef __attribute__((address_space(1))) float*        gfpw;
typedef __attribute__((address_space(1))) bf16_t*       gbfw;

__device__ __forceinline__ v16bf ld16(const bf16_t* p) {
  gv8p q = (gv8p)p;
  v8bf lo = q[0];
  v8bf hi = q[1];
  return __builtin_shufflevector(lo, hi, 0,1,2,3,4,5,6,7,8,9,10,11,12,13,14,15);
}
__device__ __forceinline__ float ldgf(const float* p)   { return *(gfp)p; }
__device__ __forceinline__ void stgf(float* p, float v) { *(gfpw)p = v; }
__device__ __forceinline__ void stgb(bf16_t* p, bf16_t v) { *(gbfw)p = v; }

// ---------------- f32 -> bf16 convert ----------------
__global__ void k_cvt(const float* __restrict__ in, bf16_t* __restrict__ out, long long n) {
  long long i = (long long)blockIdx.x * 256 + threadIdx.x;
  if (i < n) out[i] = f2bf(in[i]);
}

// ---------------- weight pack: row-major f32 [K,N] -> fragment-linear bf16 tiles ----------------
// tile (nt,kt) is 32x16 halves laid out lane-major so each lane reads 32 contiguous bytes.
// lane L holds B(k,n): n = L&15, k = kt*32 + (L>=16?16:0) + j, j=0..15
__global__ void k_pack_w(const float* __restrict__ W, bf16_t* __restrict__ out, int K, int N) {
  long long idx = (long long)blockIdx.x * 256 + threadIdx.x;
  long long total = (long long)K * N;
  if (idx >= total) return;
  int j    = (int)(idx & 15);
  int lane = (int)((idx >> 4) & 31);
  long long tl = idx >> 9;
  int Ktiles = K >> 5;
  int kt = (int)(tl % Ktiles);
  int nt = (int)(tl / Ktiles);
  int k  = kt * 32 + ((lane >> 4) << 4) + j;
  int n  = nt * 16 + (lane & 15);
  out[idx] = f2bf(W[(long long)k * N + n]);
}

// ---------------- batched pack of per-head V [Ts,HD] (bf16, row stride E) ----------------
__global__ void k_pack_v(const bf16_t* __restrict__ src, bf16_t* __restrict__ dst, int K /*Ts*/) {
  int z = blockIdx.y;                        // b*H + h
  int b = z / H_, h = z % H_;
  long long idx = (long long)blockIdx.x * 256 + threadIdx.x;   // < K*64
  int j    = (int)(idx & 15);
  int lane = (int)((idx >> 4) & 31);
  long long tl = idx >> 9;
  int Ktiles = K >> 5;
  int kt = (int)(tl % Ktiles);
  int nt = (int)(tl / Ktiles);
  int k  = kt * 32 + ((lane >> 4) << 4) + j;
  int n  = nt * 16 + (lane & 15);
  const bf16_t* s = src + (long long)b * K * E_ + (long long)h * HD_;
  dst[(long long)z * K * HD_ + idx] = s[(long long)k * E_ + n];
}

// ---------------- per-head L2 normalize of V (f32 in, bf16 out) ----------------
__global__ void k_vnorm(const float* __restrict__ Vf, bf16_t* __restrict__ Vb, long long nRowsH) {
  long long idx = (long long)blockIdx.x * 256 + threadIdx.x;   // (b*Ts)*H + h
  if (idx >= nRowsH) return;
  long long row = idx / H_;
  int h = (int)(idx % H_);
  const float* p = Vf + row * (long long)E_ + h * HD_;
  bf16_t* q = Vb + row * (long long)E_ + h * HD_;
  float ss = 0.f;
  for (int d = 0; d < HD_; ++d) { float t = ldgf(p + d); ss += t * t; }
  float inv = 1.0f / fmaxf(sqrtf(ss), 1e-12f);
  for (int d = 0; d < HD_; ++d) stgb(q + d, f2bf(ldgf(p + d) * inv));
}

// ---------------- LayerNorm over E=1024 (one block per row) ----------------
__global__ __launch_bounds__(256) void k_ln(const float* __restrict__ X,
                                            const float* __restrict__ g,
                                            const float* __restrict__ bta,
                                            float* __restrict__ outf,
                                            bf16_t* __restrict__ outb) {
  int row = blockIdx.x;
  int tid = threadIdx.x;
  const float* x = X + (long long)row * E_;
  __shared__ float s1[256], s2[256];
  float v[4], a = 0.f, b2 = 0.f;
#pragma unroll
  for (int i = 0; i < 4; ++i) { v[i] = ldgf(x + tid + i * 256); a += v[i]; b2 += v[i] * v[i]; }
  s1[tid] = a; s2[tid] = b2; __syncthreads();
  for (int s = 128; s > 0; s >>= 1) {
    if (tid < s) { s1[tid] += s1[tid + s]; s2[tid] += s2[tid + s]; }
    __syncthreads();
  }
  float mean = s1[0] * (1.0f / E_);
  float var  = s2[0] * (1.0f / E_) - mean * mean;
  float rstd = rsqrtf(var + 1e-5f);
#pragma unroll
  for (int i = 0; i < 4; ++i) {
    int c = tid + i * 256;
    float y = (v[i] - mean) * rstd * ldgf(g + c) + ldgf(bta + c);
    if (outf) stgf(outf + (long long)row * E_ + c, y);
    if (outb) stgb(outb + (long long)row * E_ + c, f2bf(y));
  }
}

// ---------------- softmax (causal-aware), f32 in -> bf16 probs ----------------
__global__ __launch_bounds__(256) void k_softmax(const float* __restrict__ S,
                                                 bf16_t* __restrict__ P,
                                                 int Tt, int Ts, int causal) {
  int t = blockIdx.x;
  long long z = blockIdx.y;
  int tid = threadIdx.x;
  const float* row = S + (z * Tt + t) * (long long)Ts;
  bf16_t* prow = P + (z * Tt + t) * (long long)Ts;
  int valid = causal ? (t + 1) : Ts;
  __shared__ float red[256];
  float mx = -3.4e38f;
  for (int j = tid; j < valid; j += 256) mx = fmaxf(mx, ldgf(row + j));
  red[tid] = mx; __syncthreads();
  for (int s = 128; s > 0; s >>= 1) { if (tid < s) red[tid] = fmaxf(red[tid], red[tid + s]); __syncthreads(); }
  mx = red[0]; __syncthreads();
  float sum = 0.f;
  for (int j = tid; j < valid; j += 256) sum += expf(ldgf(row + j) - mx);
  red[tid] = sum; __syncthreads();
  for (int s = 128; s > 0; s >>= 1) { if (tid < s) red[tid] += red[tid + s]; __syncthreads(); }
  float inv = 1.0f / red[0];
  for (int j = tid; j < Ts; j += 256)
    stgb(prow + j, f2bf(j < valid ? expf(ldgf(row + j) - mx) * inv : 0.f));
}

// ---------------- WMMA GEMM ----------------
#define GF_BIAS  1
#define GF_RESID 2
#define GF_GELU  4
#define GF_F32   8
#define GF_BF16  16

struct GemmP {
  const bf16_t* __restrict__ A;
  const bf16_t* __restrict__ Bm;
  const float*  __restrict__ bias;
  const float*  __restrict__ resid;
  float*  __restrict__ Cf;
  bf16_t* __restrict__ Cb;
  long long aB, aH, bB, bH, rB, rH, cB, cH;   // batch offsets: off = (z/H)*xB + (z%H)*xH
  int lda, ldb, ldr, ldc, K, H, bmode, flags;
  float scale;
};

// BMODE: 0 = packed fragment tiles, 1 = row-major [N,K] (B^T direct).
// WN: waves along N (1 or 2). WM = 8/WN waves along M. Each wave computes a
// 32x64 tile (2 M-tiles x 4 N-tiles, 8 accumulators); every B fragment feeds 2 WMMAs.
template <int BMODE, int WN>
__global__ __launch_bounds__(256) void k_gemm(GemmP p) {
  constexpr int WM = 8 / WN;
  const int tid   = threadIdx.x;
  const int wave  = tid >> 5;
  const int lane  = tid & 31;
  const int lhalf = lane >> 4;
  const int l15   = lane & 15;
  const int wm = wave / WN;
  const int wn = wave % WN;

  const int z  = blockIdx.z;
  const int zb = (p.H == 1) ? z : (z / p.H);
  const int zh = (p.H == 1) ? 0 : (z % p.H);
  const long long offA = (long long)zb * p.aB + (long long)zh * p.aH;
  const long long offB = (long long)zb * p.bB + (long long)zh * p.bH;
  const long long offC = (long long)zb * p.cB + (long long)zh * p.cH;
  const long long offR = (long long)zb * p.rB + (long long)zh * p.rH;

  const int mbase  = blockIdx.y * (WM * 32) + wm * 32;
  const int nbase  = blockIdx.x * (WN * 64) + wn * 64;
  const int Ktiles = p.K >> 5;

  v8f acc[2][4] = {};

  // A fragment: lane row = l15; K = kb..kb+15, kb = lhalf*16
  const bf16_t* aRow0 = p.A + offA + (long long)(mbase + l15) * p.lda + lhalf * 16;
  const bf16_t* aRow1 = aRow0 + (long long)16 * p.lda;

  // named pointers (not an array) so address-space inference stays global
  const bf16_t *b0, *b1, *b2, *b3;
  long long bStep;
  if constexpr (BMODE == 0) {
    // fragment-linear tiles: tile (ntile, kt) at ((ntile*Ktiles + kt)*32 + lane)*16
    const bf16_t* base = p.Bm + offB + (long long)lane * 16;
    const long long ntStride = (long long)Ktiles * 32 * 16;
    b0 = base + (long long)(nbase / 16 + 0) * ntStride;
    b1 = base + (long long)(nbase / 16 + 1) * ntStride;
    b2 = base + (long long)(nbase / 16 + 2) * ntStride;
    b3 = base + (long long)(nbase / 16 + 3) * ntStride;
    bStep = 32 * 16;            // next kt tile
  } else {
    // B^T direct: lane col n = nbase+nt*16+l15; 16 contiguous halves at k0 + lhalf*16
    const bf16_t* base = p.Bm + offB + (long long)(nbase + l15) * p.ldb + lhalf * 16;
    b0 = base;
    b1 = base + (long long)16 * p.ldb;
    b2 = base + (long long)32 * p.ldb;
    b3 = base + (long long)48 * p.ldb;
    bStep = 32;                 // next k0
  }

#pragma unroll 2
  for (int kt = 0; kt < Ktiles; ++kt) {
    v16bf a0 = ld16(aRow0);
    v16bf a1 = ld16(aRow1);
    v16bf bf;
    bf = ld16(b0);
    acc[0][0] = __builtin_amdgcn_wmma_f32_16x16x32_bf16(false, a0, false, bf, (short)0, acc[0][0], false, false);
    acc[1][0] = __builtin_amdgcn_wmma_f32_16x16x32_bf16(false, a1, false, bf, (short)0, acc[1][0], false, false);
    bf = ld16(b1);
    acc[0][1] = __builtin_amdgcn_wmma_f32_16x16x32_bf16(false, a0, false, bf, (short)0, acc[0][1], false, false);
    acc[1][1] = __builtin_amdgcn_wmma_f32_16x16x32_bf16(false, a1, false, bf, (short)0, acc[1][1], false, false);
    bf = ld16(b2);
    acc[0][2] = __builtin_amdgcn_wmma_f32_16x16x32_bf16(false, a0, false, bf, (short)0, acc[0][2], false, false);
    acc[1][2] = __builtin_amdgcn_wmma_f32_16x16x32_bf16(false, a1, false, bf, (short)0, acc[1][2], false, false);
    bf = ld16(b3);
    acc[0][3] = __builtin_amdgcn_wmma_f32_16x16x32_bf16(false, a0, false, bf, (short)0, acc[0][3], false, false);
    acc[1][3] = __builtin_amdgcn_wmma_f32_16x16x32_bf16(false, a1, false, bf, (short)0, acc[1][3], false, false);
    b0 += bStep; b1 += bStep; b2 += bStep; b3 += bStep;
    aRow0 += 32; aRow1 += 32;
  }

#pragma unroll
  for (int mt = 0; mt < 2; ++mt) {
#pragma unroll
    for (int nt = 0; nt < 4; ++nt) {
      const int col = nbase + nt * 16 + l15;
      const float bv = (p.flags & GF_BIAS) ? ldgf(p.bias + col) : 0.f;
#pragma unroll
      for (int r = 0; r < 8; ++r) {
        const int row = mbase + mt * 16 + r + lhalf * 8;
        float v = acc[mt][nt][r] * p.scale + bv;
        if (p.flags & GF_RESID) v += ldgf(p.resid + offR + (long long)row * p.ldr + col);
        if (p.flags & GF_GELU)  v = 0.5f * v * (1.0f + erff(v * 0.70710678118654752f));
        const long long ci = offC + (long long)row * p.ldc + col;
        if (p.flags & GF_F32)  stgf(p.Cf + ci, v);
        if (p.flags & GF_BF16) stgb(p.Cb + ci, f2bf(v));
      }
    }
  }
}

// ---------------- host orchestration ----------------
struct BP {
  const float *W1, *W2, *Wk, *Wp, *Wq, *Wv, *b1, *b2, *bk, *bp, *bq, *bv;
  const float *ln1_b, *ln1_g, *ln2_b, *ln2_g, *lnf_b, *lnf_g;
};

static BP get_bp(void* const* d_in, int i0, int L) {
  // jax tree_flatten sorted key order within a block-param dict
  static const long long per[18] = {
    (long long)E_ * 4 * E_, (long long)4 * E_ * E_,          // W1 W2
    (long long)E_ * E_, (long long)E_ * E_, (long long)E_ * E_, (long long)E_ * E_, // Wk Wp Wq Wv
    4 * E_, E_, E_, E_, E_, E_,                              // b1 b2 bk bp bq bv
    E_, E_, E_, E_, E_, E_ };                                // ln1_b ln1_g ln2_b ln2_g lnf_b lnf_g
  BP p;
  const float** f[18] = { &p.W1, &p.W2, &p.Wk, &p.Wp, &p.Wq, &p.Wv,
                          &p.b1, &p.b2, &p.bk, &p.bp, &p.bq, &p.bv,
                          &p.ln1_b, &p.ln1_g, &p.ln2_b, &p.ln2_g, &p.lnf_b, &p.lnf_g };
  for (int i = 0; i < 18; ++i)
    *f[i] = (const float*)d_in[i0 + i] + (long long)(L - 1) * per[i];
  return p;
}

// wn==2: block tile 128x128 (N%128==0). wn==1: block tile 256x64 (N==64, M%256==0).
static void launch_gemm(hipStream_t s, const GemmP& g, int M, int N, int Z, int wn) {
  if (wn == 2) {
    dim3 grid(N / 128, M / 128, Z);
    if (g.bmode == 0) k_gemm<0, 2><<<grid, dim3(256), 0, s>>>(g);
    else              k_gemm<1, 2><<<grid, dim3(256), 0, s>>>(g);
  } else {
    dim3 grid(N / 64, M / 256, Z);
    k_gemm<0, 1><<<grid, dim3(256), 0, s>>>(g);
  }
}

static void gemm_simple(hipStream_t s, const bf16_t* A, const bf16_t* Bp, const float* bias,
                        const float* resid, float* Cf, bf16_t* Cb,
                        int M, int N, int K, int flags) {
  GemmP g{};
  g.A = A; g.Bm = Bp; g.bias = bias; g.resid = resid; g.Cf = Cf; g.Cb = Cb;
  g.aB = g.aH = g.bB = g.bH = g.rB = g.rH = g.cB = g.cH = 0;
  g.lda = K; g.ldb = 0; g.ldr = N; g.ldc = N; g.K = K; g.H = 1; g.bmode = 0;
  g.flags = flags; g.scale = 1.0f;
  launch_gemm(s, g, M, N, 1, 2);
}

extern "C" void kernel_launch(void* const* d_in, const int* in_sizes, int n_in,
                              void* d_out, int out_size, void* d_ws, size_t ws_size,
                              hipStream_t stream) {
  (void)n_in; (void)out_size; (void)ws_size;
  // ---- locate inputs (two plausible flatten conventions) ----
  int iX, iHist, iStim, iBPhist, iBPstate, iBPstim, iLnfB, iLnfG;
  if (in_sizes[1] == TID_ * TID_) {
    // full sorted tree_flatten: hist, mask, params{hist(18), lnf_b, lnf_g, state(18), stim(18)}, stim, x
    iHist = 0; iBPhist = 2; iLnfB = 20; iLnfG = 21; iBPstate = 22; iBPstim = 40; iStim = 58; iX = 59;
  } else {
    // insertion order top-level, sorted params leaves: x, hist, stim, mask, params...
    iX = 0; iHist = 1; iStim = 2; iBPhist = 4; iLnfB = 22; iLnfG = 23; iBPstate = 24; iBPstim = 42;
  }
  const float* x    = (const float*)d_in[iX];
  const float* hist = (const float*)d_in[iHist];
  const float* stim = (const float*)d_in[iStim];
  const float* lnf_g = (const float*)d_in[iLnfG];
  const float* lnf_b = (const float*)d_in[iLnfB];
  BP bpS = get_bp(d_in, iBPstate, 4);
  BP bpH = get_bp(d_in, iBPhist, 2);
  BP bpT = get_bp(d_in, iBPstim, 4);

  // ---- workspace bump allocator ----
  char* w = (char*)d_ws;
  auto alloc = [&](size_t bytes) -> void* {
    void* p = (void*)w;
    w += (bytes + 255) & ~(size_t)255;
    return p;
  };
  const long long NQ = (long long)B_ * TID_;          // 4096 query rows
  const long long NKVmax = (long long)B_ * TSTIM_;    // 8192 kv rows max
  bf16_t* xbf    = (bf16_t*)alloc(NQ * E_ * 2);
  bf16_t* histbf = (bf16_t*)alloc(NQ * E_ * 2);
  bf16_t* stimbf = (bf16_t*)alloc(NKVmax * E_ * 2);
  bf16_t* wq_p = (bf16_t*)alloc((size_t)E_ * E_ * 2);
  bf16_t* wk_p = (bf16_t*)alloc((size_t)E_ * E_ * 2);
  bf16_t* wv_p = (bf16_t*)alloc((size_t)E_ * E_ * 2);
  bf16_t* wp_p = (bf16_t*)alloc((size_t)E_ * E_ * 2);
  bf16_t* w1_p = (bf16_t*)alloc((size_t)E_ * 4 * E_ * 2);
  bf16_t* w2_p = (bf16_t*)alloc((size_t)4 * E_ * E_ * 2);
  bf16_t* lnq  = (bf16_t*)alloc(NQ * E_ * 2);
  bf16_t* Qbf  = (bf16_t*)alloc(NQ * E_ * 2);
  bf16_t* Kbf  = (bf16_t*)alloc(NKVmax * E_ * 2);
  float*  Vf   = (float*)alloc(NKVmax * E_ * 4);
  bf16_t* Vbf  = (bf16_t*)alloc(NKVmax * E_ * 2);
  bf16_t* Vp   = (bf16_t*)alloc((size_t)B_ * H_ * TSTIM_ * HD_ * 2);
  float*  Sc   = (float*)alloc((size_t)B_ * H_ * TID_ * TSTIM_ * 4);   // 256 MB
  bf16_t* Pbf  = (bf16_t*)alloc((size_t)B_ * H_ * TID_ * TSTIM_ * 2);  // 128 MB
  bf16_t* Ybf  = (bf16_t*)alloc(NQ * E_ * 2);
  float*  x1f  = (float*)alloc(NQ * E_ * 4);
  bf16_t* ln2b = (bf16_t*)alloc(NQ * E_ * 2);
  bf16_t* h1bf = (bf16_t*)alloc(NQ * 4 * E_ * 2);
  float*  x2f  = (float*)alloc(NQ * E_ * 4);
  float*  outA = (float*)alloc(NQ * E_ * 4);
  float*  outB = (float*)alloc(NQ * E_ * 4);

  auto cvt = [&](const float* in, bf16_t* out, long long n) {
    k_cvt<<<dim3((unsigned)((n + 255) / 256)), dim3(256), 0, stream>>>(in, out, n);
  };
  auto packw = [&](const float* W, bf16_t* dst, int K, int N) {
    long long tot = (long long)K * N;
    k_pack_w<<<dim3((unsigned)(tot / 256)), dim3(256), 0, stream>>>(W, dst, K, N);
  };

  cvt(x, xbf, NQ * E_);
  cvt(hist, histbf, NQ * E_);
  cvt(stim, stimbf, NKVmax * E_);

  auto block = [&](const float* qf, const bf16_t* kvbf, int Ts, int causal,
                   const BP& P, float* outf) {
    const int Mq = (int)NQ;          // 4096
    const int Mkv = B_ * Ts;
    // pack weights (once per call; weights live in L2 thereafter)
    packw(P.Wq, wq_p, E_, E_); packw(P.Wk, wk_p, E_, E_); packw(P.Wv, wv_p, E_, E_);
    packw(P.Wp, wp_p, E_, E_); packw(P.W1, w1_p, E_, 4 * E_); packw(P.W2, w2_p, 4 * E_, E_);
    // ln1(q) -> bf16
    k_ln<<<dim3(Mq), dim3(256), 0, stream>>>(qf, P.ln1_g, P.ln1_b, nullptr, lnq);
    // Q = ln(q)Wq+bq ; K = kv Wk+bk ; V = kv Wv+bv
    gemm_simple(stream, lnq,  wq_p, P.bq, nullptr, nullptr, Qbf, Mq,  E_, E_, GF_BIAS | GF_BF16);
    gemm_simple(stream, kvbf, wk_p, P.bk, nullptr, nullptr, Kbf, Mkv, E_, E_, GF_BIAS | GF_BF16);
    gemm_simple(stream, kvbf, wv_p, P.bv, nullptr, Vf, nullptr, Mkv, E_, E_, GF_BIAS | GF_F32);
    // normalize V per head, pack per (b,h)
    long long nvr = (long long)Mkv * H_;
    k_vnorm<<<dim3((unsigned)(nvr / 256)), dim3(256), 0, stream>>>(Vf, Vbf, nvr);
    k_pack_v<<<dim3((unsigned)((long long)Ts * HD_ / 256), B_ * H_), dim3(256), 0, stream>>>(Vbf, Vp, Ts);
    // scores = Q K^T / 8  (batched over b,h; B^T direct mode)
    {
      GemmP g{};
      g.A = Qbf; g.Bm = Kbf; g.Cf = Sc;
      g.aB = (long long)TID_ * E_; g.aH = HD_;
      g.bB = (long long)Ts * E_;   g.bH = HD_;
      g.cB = (long long)H_ * TID_ * Ts; g.cH = (long long)TID_ * Ts;
      g.lda = E_; g.ldb = E_; g.ldr = 0; g.ldc = Ts;
      g.K = HD_; g.H = H_; g.bmode = 1; g.flags = GF_F32; g.scale = 0.125f;
      launch_gemm(stream, g, TID_, Ts, B_ * H_, 2);
    }
    k_softmax<<<dim3(TID_, B_ * H_), dim3(256), 0, stream>>>(Sc, Pbf, TID_, Ts, causal);
    // Y = P V (batched; packed-B mode), scatter heads into [B,T,E]
    {
      GemmP g{};
      g.A = Pbf; g.Bm = Vp; g.Cb = Ybf;
      g.aB = (long long)H_ * TID_ * Ts; g.aH = (long long)TID_ * Ts;
      g.bB = (long long)H_ * Ts * HD_;  g.bH = (long long)Ts * HD_;
      g.cB = (long long)TID_ * E_;      g.cH = HD_;
      g.lda = Ts; g.ldb = 0; g.ldr = 0; g.ldc = E_;
      g.K = Ts; g.H = H_; g.bmode = 0; g.flags = GF_BF16; g.scale = 1.0f;
      launch_gemm(stream, g, TID_, HD_, B_ * H_, 1);
    }
    // x1 = q + Y Wp + bp
    gemm_simple(stream, Ybf, wp_p, P.bp, qf, x1f, nullptr, Mq, E_, E_, GF_BIAS | GF_RESID | GF_F32);
    // MLP: h = gelu(ln2(x1) W1 + b1) ; x2 = x1 + h W2 + b2
    k_ln<<<dim3(Mq), dim3(256), 0, stream>>>(x1f, P.ln2_g, P.ln2_b, nullptr, ln2b);
    gemm_simple(stream, ln2b, w1_p, P.b1, nullptr, nullptr, h1bf, Mq, 4 * E_, E_, GF_BIAS | GF_GELU | GF_BF16);
    gemm_simple(stream, h1bf, w2_p, P.b2, x1f, x2f, nullptr, Mq, E_, 4 * E_, GF_BIAS | GF_RESID | GF_F32);
    // per-block final LN -> f32 (feeds next block's q)
    k_ln<<<dim3(Mq), dim3(256), 0, stream>>>(x2f, P.lnf_g, P.lnf_b, outf, nullptr);
  };

  // Only the LAST block of each stack survives in the reference (q,k,v are re-used,
  // x is overwritten) — execute exactly those three blocks.
  block(x,    xbf,    TID_,   1, bpS, outA);   // state: causal self-attn
  block(outA, histbf, TID_,   0, bpH, outB);   // hist: cross-attn
  block(outB, stimbf, TSTIM_, 0, bpT, outA);   // stim: cross-attn
  // overall final LayerNorm -> d_out (f32)
  k_ln<<<dim3((int)NQ), dim3(256), 0, stream>>>(outA, lnf_g, lnf_b, (float*)d_out, nullptr);
}
